// SimpleLSTMCaptioningModel_79577154060753
// MI455X (gfx1250) — compile-verified, matrix operand order checked by reading, not run on hardware
//
#include <hip/hip_runtime.h>
#include <hip/hip_bf16.h>

typedef __bf16 bf16_t;
typedef __attribute__((ext_vector_type(16))) __bf16 v16bf;
typedef __attribute__((ext_vector_type(8)))  __bf16 v8bf;
typedef __attribute__((ext_vector_type(8)))  float  v8f;

#define B_  128
#define T_  35
#define V_  10000
#define F_  1408
#define E_  512
#define H_  512
#define R_  81
#define G4  2048   // 4*H
#define EF  1920   // E+F

// ---- WMMA fragment helpers (bf16 A/B row-major-K, f32 C/D) ----------------
// A 16x32 frag: lane L holds row M = L&15; K = koff..koff+7 and koff+16..+23,
// koff = (L>>4)*8.  B 32x16 frag mirrors with N in place of M.  Both our A
// (activations, MxK row-major) and W (NxK row-major) match this directly.
// Works for global or LDS pointers (generic address space -> ds_load_b128
// when p points into __shared__).
__device__ __forceinline__ v16bf load_frag(const bf16_t* p) {
  v8bf lo = *(const v8bf*)p;
  v8bf hi = *(const v8bf*)(p + 16);
  v16bf f;
#pragma unroll
  for (int i = 0; i < 8; ++i) { f[i] = lo[i]; f[i + 8] = hi[i]; }
  return f;
}

__device__ __forceinline__ v8f wmma_acc(v16bf a, v16bf b, v8f c) {
  return __builtin_amdgcn_wmma_f32_16x16x32_bf16(false, a, false, b, (short)0, c,
                                                 false, false);
}

// K-loop over 32-wide chunks; aptr/bptr are already per-lane offset.
__device__ __forceinline__ v8f gemm_tile(const bf16_t* aptr, const bf16_t* bptr,
                                         int kchunks, v8f acc) {
#pragma unroll 4
  for (int k = 0; k < kchunks; ++k) {
    v16bf af = load_frag(aptr + k * 32);
    v16bf bf = load_frag(bptr + k * 32);
    acc = wmma_acc(af, bf, acc);
  }
  return acc;
}

// Fast activations: one v_exp_f32 + one v_rcp_f32 each (no IEEE div
// expansion); saturate correctly at +-inf.
__device__ __forceinline__ float rcp_(float x) {
  return __builtin_amdgcn_rcpf(x);
}
__device__ __forceinline__ float sigmoidf_(float x) {
  return rcp_(1.0f + __expf(-x));
}
__device__ __forceinline__ float tanhf_(float x) {
  return 1.0f - 2.0f * rcp_(__expf(2.0f * x) + 1.0f);
}

// ---- 0: mean over R regions + bf16 copy -----------------------------------
__global__ void k_mean(const float* __restrict__ img, float* __restrict__ mf,
                       bf16_t* __restrict__ mbf) {
  int i = blockIdx.x * blockDim.x + threadIdx.x;
  if (i >= B_ * F_) return;
  int b = i / F_, f = i - b * F_;
  const float* p = img + (size_t)b * R_ * F_ + f;
  float s = 0.f;
  for (int r = 0; r < R_; ++r) s += p[(size_t)r * F_];
  s *= (1.0f / R_);
  mf[i] = s;
  mbf[i] = (bf16_t)s;
}

// ---- 1: fp32 -> bf16 weight conversion ------------------------------------
__global__ void k_cvt(const float* __restrict__ src, bf16_t* __restrict__ dst,
                      int n) {
  int i = blockIdx.x * blockDim.x + threadIdx.x;
  if (i < n) dst[i] = (bf16_t)src[i];
}

// ---- 2: h0 = tanh(mean @ Wihᵀ + b), c0 likewise.  512 tiles, 1 tile/wave --
__global__ void k_init_hc(const bf16_t* __restrict__ mbf,
                          const bf16_t* __restrict__ Wh, const float* __restrict__ bh,
                          const bf16_t* __restrict__ Wc, const float* __restrict__ bc,
                          bf16_t* __restrict__ h0, float* __restrict__ c0) {
  int wave = threadIdx.x >> 5, lane = threadIdx.x & 31;
  int tile = blockIdx.x * 8 + wave;          // 0..511
  int sel  = tile >> 8;                      // 0: h, 1: c
  int mt   = (tile & 255) >> 5, nt = tile & 31;
  int mbase = mt * 16, nbase = nt * 16;
  int koff  = (lane >> 4) << 3;
  const bf16_t* W    = sel ? Wc : Wh;
  const float*  bias = sel ? bc : bh;
  const bf16_t* aptr = mbf + (mbase + (lane & 15)) * F_ + koff;
  const bf16_t* bptr = W + (size_t)(nbase + (lane & 15)) * F_ + koff;
  v8f acc = {};
  acc = gemm_tile(aptr, bptr, F_ / 32, acc);
  int col = nbase + (lane & 15);
  int rbase = mbase + ((lane >> 4) << 3);
  float bv = bias[col];
#pragma unroll
  for (int r = 0; r < 8; ++r) {
    float v = tanhf_(acc[r] + bv);
    if (sel) c0[(rbase + r) * H_ + col] = v;
    else     h0[(rbase + r) * H_ + col] = (bf16_t)v;
  }
}

// ---- 3: gmean = mean @ W_ih[:,E:]ᵀ + b_ih + b_hh  (128x2048, K=1408) ------
__global__ void k_gmean(const bf16_t* __restrict__ mbf,
                        const bf16_t* __restrict__ Wih,
                        const float* __restrict__ bih,
                        const float* __restrict__ bhh,
                        float* __restrict__ gmean) {
  int wave = threadIdx.x >> 5, lane = threadIdx.x & 31;
  int tile = blockIdx.x * 8 + wave;          // 0..1023 (8 x 128)
  int mt = tile >> 7, nt = tile & 127;
  int mbase = mt * 16, nbase = nt * 16;
  int koff = (lane >> 4) << 3;
  const bf16_t* aptr = mbf + (mbase + (lane & 15)) * F_ + koff;
  const bf16_t* bptr = Wih + (size_t)(nbase + (lane & 15)) * EF + E_ + koff;
  v8f acc = {};
  acc = gemm_tile(aptr, bptr, F_ / 32, acc);
  int col = nbase + (lane & 15);
  int rbase = mbase + ((lane >> 4) << 3);
  float bv = bih[col] + bhh[col];
#pragma unroll
  for (int r = 0; r < 8; ++r)
    gmean[(rbase + r) * G4 + col] = acc[r] + bv;
}

// ---- 4: gx[rt] = emb[seq[rt]] @ W_ih[:,:E]ᵀ + gmean[b]  (4480x2048) -------
__global__ void k_gx(const bf16_t* __restrict__ emb, const int* __restrict__ seq,
                     const bf16_t* __restrict__ Wih,
                     const float* __restrict__ gmean, float* __restrict__ gx) {
  int wave = threadIdx.x >> 5, lane = threadIdx.x & 31;
  int tile = blockIdx.x * 8 + wave;          // 0..35839 (280 x 128)
  int mt = tile >> 7, nt = tile & 127;
  int mbase = mt * 16, nbase = nt * 16;
  int koff = (lane >> 4) << 3;
  int row = mbase + (lane & 15);             // rt = b*T + t
  int vocab = seq[row];                      // per-lane row gather
  const bf16_t* aptr = emb + (size_t)vocab * E_ + koff;
  const bf16_t* bptr = Wih + (size_t)(nbase + (lane & 15)) * EF + koff;
  v8f acc = {};
  acc = gemm_tile(aptr, bptr, E_ / 32, acc);
  int col = nbase + (lane & 15);
  int rbase = mbase + ((lane >> 4) << 3);
#pragma unroll
  for (int r = 0; r < 8; ++r) {
    int rt = rbase + r;
    int b = rt / T_;
    gx[(size_t)rt * G4 + col] = acc[r] + gmean[b * G4 + col];
  }
}

// ---- 5: one LSTM step, fully fused.  Block = 4 waves, wave q computes the
// 16x16 tile of gate quadrant q (gates = gx_t + h_prev @ W_hhᵀ); quadrants
// meet in LDS; cell update + h/c write in the same kernel. -------------------
__global__ void k_step(const float* __restrict__ gx,
                       const bf16_t* __restrict__ Whh,
                       const bf16_t* __restrict__ hprev,
                       const float* __restrict__ cprev,
                       bf16_t* __restrict__ hnext, float* __restrict__ cnext,
                       bf16_t* __restrict__ hseq, int t) {
  __shared__ float gsh[4][16][16];
  int q = threadIdx.x >> 5, lane = threadIdx.x & 31;
  int mbase = blockIdx.x * 16;               // batch rows
  int nbase = blockIdx.y * 16;               // hidden cols
  int gcol = q * H_ + nbase;
  int koff = (lane >> 4) << 3;
  const bf16_t* aptr = hprev + (mbase + (lane & 15)) * H_ + koff;
  const bf16_t* bptr = Whh + (size_t)(gcol + (lane & 15)) * H_ + koff;
  v8f acc = {};
  acc = gemm_tile(aptr, bptr, H_ / 32, acc);
  int nl = lane & 15;
  int rlbase = (lane >> 4) << 3;
  int colg = gcol + nl;
#pragma unroll
  for (int r = 0; r < 8; ++r) {
    int b = mbase + rlbase + r;
    gsh[q][rlbase + r][nl] = acc[r] + gx[((size_t)b * T_ + t) * G4 + colg];
  }
  __syncthreads();
#pragma unroll
  for (int e = 0; e < 2; ++e) {
    int idx = threadIdx.x * 2 + e;           // 0..255 -> 16x16 cell updates
    int ml = idx >> 4, ncl = idx & 15;
    float iv = sigmoidf_(gsh[0][ml][ncl]);
    float fv = sigmoidf_(gsh[1][ml][ncl]);
    float gv = tanhf_(gsh[2][ml][ncl]);
    float ov = sigmoidf_(gsh[3][ml][ncl]);
    int b = mbase + ml, col = nbase + ncl;
    float c = fv * cprev[b * H_ + col] + iv * gv;
    float h = ov * tanhf_(c);
    cnext[b * H_ + col] = c;
    bf16_t hb = (bf16_t)h;
    hnext[b * H_ + col] = hb;
    hseq[((size_t)b * T_ + t) * H_ + col] = hb;
  }
}

// ---- 6: out_seq = relu(h_seq @ W_fc1ᵀ + b)  (4480x512, K=512) -------------
__global__ void k_fc1(const bf16_t* __restrict__ hseq,
                      const bf16_t* __restrict__ W,
                      const float* __restrict__ bias,
                      bf16_t* __restrict__ outseq) {
  int wave = threadIdx.x >> 5, lane = threadIdx.x & 31;
  int tile = blockIdx.x * 8 + wave;          // 0..8959 (280 x 32)
  int mt = tile >> 5, nt = tile & 31;
  int mbase = mt * 16, nbase = nt * 16;
  int koff = (lane >> 4) << 3;
  const bf16_t* aptr = hseq + (size_t)(mbase + (lane & 15)) * H_ + koff;
  const bf16_t* bptr = W + (size_t)(nbase + (lane & 15)) * H_ + koff;
  v8f acc = {};
  acc = gemm_tile(aptr, bptr, H_ / 32, acc);
  int col = nbase + (lane & 15);
  int rbase = mbase + ((lane >> 4) << 3);
  float bv = bias[col];
#pragma unroll
  for (int r = 0; r < 8; ++r) {
    float v = acc[r] + bv;
    outseq[(size_t)(rbase + r) * H_ + col] = (bf16_t)(v > 0.f ? v : 0.f);
  }
}

// ---- 7: logits = out_seq @ W_fc2ᵀ + b  (4480x10000, K=512) ----------------
// Block = 8 waves (2x4), block tile 64x128, K staged through LDS in 32-wide
// chunks with double buffering (24 KB of the 320 KB WGP LDS).  Each wave
// holds 2x2 f32 accumulators (32x32) and reads fragments via ds_load_b128.
#define FC2_KC (H_ / 32)
__global__ void k_fc2(const bf16_t* __restrict__ outseq,
                      const bf16_t* __restrict__ W,
                      const float* __restrict__ bias, float* __restrict__ out) {
  __shared__ bf16_t Asm[2][64 * 32];    // 4 KB per buffer
  __shared__ bf16_t Bsm[2][128 * 32];   // 8 KB per buffer
  int tid = threadIdx.x;
  int wave = tid >> 5, lane = tid & 31;
  int wm = wave >> 2, wn = wave & 3;
  int mblock = blockIdx.y * 64;              // M = 4480 exact
  int nblock = blockIdx.x * 128;             // N = 10000, edge clamped
  int l15 = lane & 15;
  int koff = (lane >> 4) << 3;

  // cooperative stage of one 32-wide K chunk into LDS
  auto stage = [&](int k, int buf) {
    int kb = k * 32;
    {
      int row = tid >> 2, kp = (tid & 3) << 3;               // 64x32: 1x b128/thr
      *(v8bf*)&Asm[buf][row * 32 + kp] =
          *(const v8bf*)(outseq + (size_t)(mblock + row) * H_ + kb + kp);
    }
#pragma unroll
    for (int h = 0; h < 2; ++h) {                            // 128x32: 2x b128/thr
      int idx = tid + h * 256;
      int col = idx >> 2, kp = (idx & 3) << 3;
      int gc = nblock + col;
      gc = gc < V_ ? gc : V_ - 1;                            // clamp edge loads
      *(v8bf*)&Bsm[buf][col * 32 + kp] =
          *(const v8bf*)(W + (size_t)gc * H_ + kb + kp);
    }
  };

  v8f acc00 = {}, acc01 = {}, acc10 = {}, acc11 = {};
  stage(0, 0);
#pragma unroll 2
  for (int k = 0; k < FC2_KC; ++k) {
    int buf = k & 1;
    __syncthreads();
    if (k + 1 < FC2_KC) stage(k + 1, buf ^ 1);
    const bf16_t* As = &Asm[buf][0];
    const bf16_t* Bs = &Bsm[buf][0];
    v16bf af0 = load_frag(As + (wm * 32 + l15) * 32 + koff);
    v16bf af1 = load_frag(As + (wm * 32 + 16 + l15) * 32 + koff);
    v16bf bf0 = load_frag(Bs + (wn * 32 + l15) * 32 + koff);
    v16bf bf1 = load_frag(Bs + (wn * 32 + 16 + l15) * 32 + koff);
    acc00 = wmma_acc(af0, bf0, acc00);
    acc01 = wmma_acc(af0, bf1, acc01);
    acc10 = wmma_acc(af1, bf0, acc10);
    acc11 = wmma_acc(af1, bf1, acc11);
  }

  int rb = (lane >> 4) << 3;
  int mbase = mblock + wm * 32;
  int nbase = nblock + wn * 32;
  auto store_t = [&](v8f acc, int mb, int nb) {
    int col = nb + l15;
    if (col >= V_) return;                   // uniform: V_ % 16 == 0
    float bv = bias[col];
    int rbase = mb + rb;
#pragma unroll
    for (int r = 0; r < 8; ++r)
      out[(size_t)(rbase + r) * V_ + col] = acc[r] + bv;
  };
  store_t(acc00, mbase,      nbase);
  store_t(acc01, mbase,      nbase + 16);
  store_t(acc10, mbase + 16, nbase);
  store_t(acc11, mbase + 16, nbase + 16);
}

// ---------------------------------------------------------------------------
extern "C" void kernel_launch(void* const* d_in, const int* in_sizes, int n_in,
                              void* d_out, int out_size, void* d_ws, size_t ws_size,
                              hipStream_t stream) {
  const float* img     = (const float*)d_in[0];
  const int*   seq     = (const int*)d_in[1];
  // d_in[2] = lengths (unused: logits computed for all t, as in the reference)
  const float* emb     = (const float*)d_in[3];
  const float* W_ih    = (const float*)d_in[4];
  const float* b_ih    = (const float*)d_in[5];
  const float* W_hh    = (const float*)d_in[6];
  const float* b_hh    = (const float*)d_in[7];
  const float* W_inh   = (const float*)d_in[8];
  const float* b_inh   = (const float*)d_in[9];
  const float* W_inc   = (const float*)d_in[10];
  const float* b_inc   = (const float*)d_in[11];
  const float* W_fc1   = (const float*)d_in[12];
  const float* b_fc1   = (const float*)d_in[13];
  const float* W_fc2   = (const float*)d_in[14];
  const float* b_fc2   = (const float*)d_in[15];
  float* out = (float*)d_out;

  char* p = (char*)d_ws;
  auto alloc = [&](size_t bytes) {
    void* r = (void*)p;
    p += (bytes + 255) & ~(size_t)255;
    return r;
  };
  bf16_t* emb_bf  = (bf16_t*)alloc((size_t)V_ * E_ * 2);
  bf16_t* Wih_bf  = (bf16_t*)alloc((size_t)G4 * EF * 2);
  bf16_t* Whh_bf  = (bf16_t*)alloc((size_t)G4 * H_ * 2);
  bf16_t* Wfc1_bf = (bf16_t*)alloc((size_t)H_ * H_ * 2);
  bf16_t* Wfc2_bf = (bf16_t*)alloc((size_t)V_ * H_ * 2);
  bf16_t* Winh_bf = (bf16_t*)alloc((size_t)H_ * F_ * 2);
  bf16_t* Winc_bf = (bf16_t*)alloc((size_t)H_ * F_ * 2);
  float*  mean_f  = (float*)alloc((size_t)B_ * F_ * 4);
  bf16_t* mean_bf = (bf16_t*)alloc((size_t)B_ * F_ * 2);
  float*  gmean   = (float*)alloc((size_t)B_ * G4 * 4);
  float*  gx      = (float*)alloc((size_t)B_ * T_ * G4 * 4);
  bf16_t* hbuf0   = (bf16_t*)alloc((size_t)B_ * H_ * 2);
  bf16_t* hbuf1   = (bf16_t*)alloc((size_t)B_ * H_ * 2);
  float*  cbuf0   = (float*)alloc((size_t)B_ * H_ * 4);
  float*  cbuf1   = (float*)alloc((size_t)B_ * H_ * 4);
  bf16_t* hseq    = (bf16_t*)alloc((size_t)B_ * T_ * H_ * 2);
  bf16_t* outseq  = (bf16_t*)alloc((size_t)B_ * T_ * H_ * 2);
  bf16_t* hb[2] = {hbuf0, hbuf1};
  float*  cb[2] = {cbuf0, cbuf1};

  // Stage 0: feature mean + weight conversion (bandwidth-trivial, ~170 MB)
  k_mean<<<(B_ * F_ + 255) / 256, 256, 0, stream>>>(img, mean_f, mean_bf);
  auto cvt = [&](const float* s, bf16_t* d, int n) {
    k_cvt<<<(n + 255) / 256, 256, 0, stream>>>(s, d, n);
  };
  cvt(emb,   emb_bf,  V_ * E_);
  cvt(W_ih,  Wih_bf,  G4 * EF);
  cvt(W_hh,  Whh_bf,  G4 * H_);
  cvt(W_fc1, Wfc1_bf, H_ * H_);
  cvt(W_fc2, Wfc2_bf, V_ * H_);
  cvt(W_inh, Winh_bf, H_ * F_);
  cvt(W_inc, Winc_bf, H_ * F_);

  // Stage 1: h0/c0 and time-invariant gate contributions
  k_init_hc<<<64, 256, 0, stream>>>(mean_bf, Winh_bf, b_inh, Winc_bf, b_inc,
                                    hb[0], cb[0]);
  k_gmean<<<128, 256, 0, stream>>>(mean_bf, Wih_bf, b_ih, b_hh, gmean);
  k_gx<<<4480, 256, 0, stream>>>(emb_bf, seq, Wih_bf, gmean, gx);

  // Stage 2: sequential recurrence, one fused kernel per step
  for (int t = 0; t < T_; ++t)
    k_step<<<dim3(8, 32), 128, 0, stream>>>(gx, Whh_bf, hb[t & 1], cb[t & 1],
                                            hb[(t + 1) & 1], cb[(t + 1) & 1],
                                            hseq, t);

  // Stage 3: batched fc1 + fc2 over all (b,t) rows (the 46-GFLOP GEMM)
  k_fc1<<<1120, 256, 0, stream>>>(hseq, Wfc1_bf, b_fc1, outseq);
  k_fc2<<<dim3(79, 70), 256, 0, stream>>>(outseq, Wfc2_bf, b_fc2, out);
}
